// SelfAttentionNetWeighted_Factor_17583596110394
// MI455X (gfx1250) — matrix operand order repaired and checked by reading, not compiled, not run on hardware
//
#include <hip/hip_runtime.h>
#include <hip/hip_bf16.h>
#include <math.h>

typedef __attribute__((ext_vector_type(16))) _Float16 v16h;
typedef __attribute__((ext_vector_type(8)))  float    v8f;

#define S_DIM 1200
#define XROW  1776          // S + E*T + E*4
#define BUS   300
#define BA_N  1024
#define NTOT  (BA_N * BUS)  // 307200
#define EPSC  1e-5f

// ---- WMMA fragment helpers (wave32, CDNA5 16x16x32 f16 layouts, ISA 7.12.2) ----
// A: 16x32 (MxK) f16, row-major in LDS. Two contiguous 16B chunks per lane -> 2x ds_load_b128.
__device__ inline v16h wmma_load_a(const _Float16* src, int ld, int lane) {
  v16h a;
  const int row  = lane & 15;
  const int kofs = (lane < 16) ? 0 : 16;
  const _Float16* p = src + row * ld + kofs;
#pragma unroll
  for (int h = 0; h < 8; ++h) a[h] = p[h];
#pragma unroll
  for (int h = 0; h < 8; ++h) a[8 + h] = p[8 + h];
  return a;
}

// B fragments are staged PRE-SWIZZLED: for tile t (32x16 KxN), lane l's 16 halves
// live contiguously at dst[t*512 + l*16 + h], where l = (kk>=16)*16 + col, h = kk&15.
// Fragment load is then one aligned v16h LDS read (2x ds_load_b128, no packing).
template <int K, int N, int KREAL>
__device__ inline void stage_b(_Float16* __restrict__ dst, const float* __restrict__ src,
                               int tid, int nthreads) {
  constexpr int NT = N / 16;
  for (int i = tid; i < K * N; i += nthreads) {
    int tile   = i >> 9;
    int within = i & 511;
    int lw  = within >> 4;
    int h   = within & 15;
    int kt  = tile / NT;
    int nt  = tile - kt * NT;
    int kk  = ((lw >> 4) << 4) | h;
    int col = lw & 15;
    int k   = kt * 32 + kk;
    int n   = nt * 16 + col;
    float v = (k < KREAL) ? src[k * N + n] : 0.f;
    dst[i] = (_Float16)v;
  }
}

__device__ inline v16h wmma_load_b_sw(const _Float16* sh, int tile, int lane) {
  return *(const v16h*)(sh + tile * 512 + lane * 16);
}

__device__ inline v8f wmma_f16(v16h a, v16h b, v8f c) {
  return __builtin_amdgcn_wmma_f32_16x16x32_f16(false, a, false, b, (short)0, c,
                                                false, false);
}

// ---- Kernel 1: per-block partial sums/sumsq of the 4 state phases (deterministic) ----
__global__ void k_stats_partial(const float* __restrict__ x, float* __restrict__ part) {
  const int tid = threadIdx.x;
  __shared__ float red[256][8];
  float s[4] = {0.f, 0.f, 0.f, 0.f}, q[4] = {0.f, 0.f, 0.f, 0.f};
  const int total  = NTOT * 4;
  const int stride = blockDim.x * gridDim.x;
  for (int i = blockIdx.x * blockDim.x + tid; i < total; i += stride) {
    int b = i / S_DIM, sidx = i - b * S_DIM;
    float v = x[(size_t)b * XROW + sidx];
    int j = sidx & 3;
    s[j] += v;
    q[j] += v * v;
  }
#pragma unroll
  for (int j = 0; j < 4; ++j) { red[tid][j] = s[j]; red[tid][4 + j] = q[j]; }
  __syncthreads();
  for (int off = 128; off > 0; off >>= 1) {
    if (tid < off)
#pragma unroll
      for (int j = 0; j < 8; ++j) red[tid][j] += red[tid + off][j];
    __syncthreads();
  }
  if (tid < 8) part[blockIdx.x * 8 + tid] = red[0][tid];
}

__global__ void k_stats_final(const float* __restrict__ part, float* __restrict__ stats,
                              int nblk) {
  int tid = threadIdx.x;
  if (tid < 8) {
    float s = 0.f;
    for (int i = 0; i < nblk; ++i) s += part[i * 8 + tid];
    stats[tid] = s;
  }
}

// ---- Kernel 2: qv[b,:] = softmax(alpha) . MLP(emb_task)  (WMMA 2-layer MLP) ----
__global__ void k_qv(const float* __restrict__ x, const float* __restrict__ wq_W1,
                     const float* __restrict__ wq_b1, const float* __restrict__ wq_W2,
                     const float* __restrict__ wq_b2, const float* __restrict__ alpha,
                     float* __restrict__ qv) {
  const int b = blockIdx.x;
  const int tid = threadIdx.x;   // 64 threads = 2 waves
  const int lane = tid & 31, wave = tid >> 5;
  __shared__ __align__(32) _Float16 sh_et[32 * 64];    // emb_task, K padded 48->64 (A, row-major)
  __shared__ __align__(32) _Float16 sh_w1[64 * 128];   // wq_W1 swizzled B tiles
  __shared__ __align__(32) _Float16 sh_h1[32 * 128];   // h1 (A, row-major)
  __shared__ __align__(32) _Float16 sh_w2[128 * 64];   // wq_W2 swizzled B tiles
  __shared__ float    sh_q[32 * 64];
  __shared__ float    sh_an[32];

  const float* xb = x + (size_t)b * XROW;
  for (int i = tid; i < 32 * 64; i += 64) {
    int r = i >> 6, k = i & 63;
    float v = 0.f;
    if (k < 48) {
      int t = r >> 2, j = r & 3;
      v = xb[1200 + t * 48 + k] * xb[1584 + j * 48 + k];
    }
    sh_et[i] = (_Float16)v;
  }
  stage_b<64, 128, 48>(sh_w1, wq_W1, tid, 64);
  stage_b<128, 64, 128>(sh_w2, wq_W2, tid, 64);
  if (tid == 0) {
    float mx = alpha[0];
    for (int i = 1; i < 32; ++i) mx = fmaxf(mx, alpha[i]);
    float ssum = 0.f;
    for (int i = 0; i < 32; ++i) { float e = expf(alpha[i] - mx); sh_an[i] = e; ssum += e; }
    float inv = 1.f / ssum;
    for (int i = 0; i < 32; ++i) sh_an[i] *= inv;
  }
  __syncthreads();

  const int row0 = wave * 16;
  const int col15 = lane & 15;
  const int rsel = (lane < 16) ? 0 : 8;
  // layer 1: h1 = relu(et @ W1 + b1)    B tiles: (k0/32)*8 + nt
  for (int nt = 0; nt < 8; ++nt) {
    v8f acc = {};
#pragma unroll
    for (int kt = 0; kt < 2; ++kt) {
      v16h a  = wmma_load_a(&sh_et[row0 * 64 + kt * 32], 64, lane);
      v16h bm = wmma_load_b_sw(sh_w1, kt * 8 + nt, lane);
      acc = wmma_f16(a, bm, acc);
    }
    int colp = nt * 16 + col15;
#pragma unroll
    for (int i = 0; i < 8; ++i)
      sh_h1[(row0 + rsel + i) * 128 + colp] = (_Float16)fmaxf(acc[i] + wq_b1[colp], 0.f);
  }
  __syncthreads();
  // layer 2: q = h1 @ W2 + b2          B tiles: (k0/32)*4 + nt
  for (int nt = 0; nt < 4; ++nt) {
    v8f acc = {};
#pragma unroll
    for (int kt = 0; kt < 4; ++kt) {
      v16h a  = wmma_load_a(&sh_h1[row0 * 128 + kt * 32], 128, lane);
      v16h bm = wmma_load_b_sw(sh_w2, kt * 4 + nt, lane);
      acc = wmma_f16(a, bm, acc);
    }
    int colp = nt * 16 + col15;
#pragma unroll
    for (int i = 0; i < 8; ++i)
      sh_q[(row0 + rsel + i) * 64 + colp] = acc[i] + wq_b2[colp];
  }
  __syncthreads();
  if (tid < 64) {
    float s = 0.f;
    for (int r = 0; r < 32; ++r) s += sh_an[r] * sh_q[r * 64 + tid];
    qv[b * 64 + tid] = s;
  }
}

// ---- Kernel 3: per-(f,b) attention over 300 bus slots (algebraically reduced) ----
__global__ void k_attn(const float* __restrict__ x, const float* __restrict__ tw_W1,
                       const float* __restrict__ tw_b1, const float* __restrict__ tw_gamma,
                       const float* __restrict__ tw_beta, const float* __restrict__ tw_W2,
                       const float* __restrict__ tw_b2, const float* __restrict__ stats,
                       const float* __restrict__ qv, float* __restrict__ rws) {
  const int b = blockIdx.x;
  const int f = blockIdx.y;        // factor pair 0..3
  const int tid = threadIdx.x;     // 256
  __shared__ float4 kpk[128];      // {a_k, d_k, w2q, 0}
  __shared__ float2 vpk[128];      // {a_v, d_v}
  __shared__ float vbuf[BUS], sc[BUS];
  __shared__ float red[256];
  __shared__ float sh_cb;

  const float invN = 1.f / (float)NTOT;
  const float m  = stats[f] * invN;
  const float vx = fmaxf(stats[4 + f] * invN - m * m, 0.f);

  {
    int c = tid & 127;
    int fac = 2 * f + (tid >> 7);
    float W  = tw_W1[fac * 128 + c];
    float a  = W * rsqrtf(W * W * vx + EPSC) * tw_gamma[fac * 128 + c];
    float d0 = tw_beta[fac * 128 + c] - a * m;
    if (tid < 128) { kpk[c].x = a; kpk[c].y = d0; kpk[c].w = 0.f; }
    else           { vpk[c].x = a; vpk[c].y = d0; }
  }
  if (tid < 128) {   // fold key-side W2 into qv
    float s = 0.f;
    const float* w2 = tw_W2 + (size_t)(2 * f) * 128 * 64 + tid * 64;
    const float* q  = qv + b * 64;
    for (int d = 0; d < 64; ++d) s += w2[d] * q[d];
    kpk[tid].z = s;
  }
  if (tid == 0) {
    float s = 0.f;
    const float* bb = tw_b2 + (2 * f) * 64;
    const float* q  = qv + b * 64;
    for (int d = 0; d < 64; ++d) s += bb[d] * q[d];
    sh_cb = s;
  }
  for (int n = tid; n < BUS; n += 256) vbuf[n] = x[(size_t)b * XROW + 4 * n + f];
  __syncthreads();

  float lmax = -1e30f;
  for (int n = tid; n < BUS; n += 256) {
    float v = vbuf[n];
    float s = sh_cb;
#pragma unroll 4
    for (int c = 0; c < 128; ++c) {
      float4 p = kpk[c];
      s += fmaxf(fmaf(p.x, v, p.y), 0.f) * p.z;
    }
    s *= 0.125f;                 // 1/sqrt(64)
    sc[n] = s;
    lmax = fmaxf(lmax, s);
  }
  red[tid] = lmax; __syncthreads();
  for (int off = 128; off > 0; off >>= 1) {
    if (tid < off) red[tid] = fmaxf(red[tid], red[tid + off]);
    __syncthreads();
  }
  const float mx = red[0];
  float lsum = 0.f;
  for (int n = tid; n < BUS; n += 256) {
    float e = expf(sc[n] - mx);
    sc[n] = e;
    lsum += e;
  }
  __syncthreads();
  red[tid] = lsum; __syncthreads();
  for (int off = 128; off > 0; off >>= 1) {
    if (tid < off) red[tid] += red[tid + off];
    __syncthreads();
  }
  const float inv_sum = 1.f / red[0];

  const int c = tid & 127, half = tid >> 7;
  const float2 pv = vpk[c];
  float acc = 0.f;
  const int n0 = half * 150, n1 = n0 + 150;
  for (int n = n0; n < n1; ++n)
    acc += sc[n] * fmaxf(fmaf(pv.x, vbuf[n], pv.y), 0.f);
  __syncthreads();
  red[tid] = acc; __syncthreads();
  if (tid < 128)
    rws[((size_t)b * 4 + f) * 128 + c] = (red[tid] + red[tid + 128]) * inv_sum;
}

// ---- Kernel 4: hsum GEMM + Q/V heads + mean fixup (WMMA chain, 16 rows/block) ----
__global__ void k_head(const float* __restrict__ rws, const float* __restrict__ tw_W2,
                       const float* __restrict__ tw_b2, const float* __restrict__ Q_W1,
                       const float* __restrict__ Q_b1, const float* __restrict__ Q_W2,
                       const float* __restrict__ Q_b2, const float* __restrict__ V_W1,
                       const float* __restrict__ V_b1, const float* __restrict__ V_W2,
                       const float* __restrict__ V_b2, float* __restrict__ out) {
  const int b0 = blockIdx.x * 16;
  const int lane = threadIdx.x;     // 32 threads = 1 wave
  __shared__ __align__(32) _Float16 sh_r[16 * 512];   // A, row-major
  __shared__ __align__(32) _Float16 sh_w[64 * 128];   // swizzled B staging (16 KB, reused)
  __shared__ __align__(32) _Float16 sh_h[16 * 64];    // hsum (A)
  __shared__ __align__(32) _Float16 sh_a[16 * 128];   // t1 (A)
  __shared__ float    sh_qo[16 * 128];
  __shared__ float    sh_vt[16 * 128];
  __shared__ float    sh_vo[16], sh_qm[16];

  for (int i = lane; i < 16 * 512; i += 32)
    sh_r[i] = (_Float16)rws[(size_t)(b0 + (i >> 9)) * 512 + (i & 511)];

  const int col15 = lane & 15;
  const int rsel = (lane < 16) ? 0 : 8;
  const v8f vzero = {};

  // hsum = r(16x512) @ Wv(512x64) + sum_f b2v   (K chunked by factor; B tiles (k0/32)*4+nt)
  v8f acc4[4];
#pragma unroll
  for (int nt = 0; nt < 4; ++nt) acc4[nt] = vzero;
  for (int kc = 0; kc < 4; ++kc) {
    __syncthreads();
    stage_b<128, 64, 128>(sh_w, tw_W2 + (size_t)(2 * kc + 1) * 8192, lane, 32);
    __syncthreads();
    for (int nt = 0; nt < 4; ++nt)
#pragma unroll
      for (int kt = 0; kt < 4; ++kt) {
        v16h a  = wmma_load_a(&sh_r[kc * 128 + kt * 32], 512, lane);
        v16h bm = wmma_load_b_sw(sh_w, kt * 4 + nt, lane);
        acc4[nt] = wmma_f16(a, bm, acc4[nt]);
      }
  }
  for (int nt = 0; nt < 4; ++nt) {
    int colp = nt * 16 + col15;
    float bias = tw_b2[64 + colp] + tw_b2[192 + colp] + tw_b2[320 + colp] + tw_b2[448 + colp];
#pragma unroll
    for (int i = 0; i < 8; ++i)
      sh_h[(rsel + i) * 64 + colp] = (_Float16)(acc4[nt][i] + bias);
  }
  __syncthreads();

  // Q head layer 1: t1 = relu(hsum @ Q_W1 + Q_b1)   (64x128, B tiles (k0/32)*8+nt)
  stage_b<64, 128, 64>(sh_w, Q_W1, lane, 32);
  __syncthreads();
  for (int nt = 0; nt < 8; ++nt) {
    v8f acc = vzero;
#pragma unroll
    for (int kt = 0; kt < 2; ++kt) {
      v16h a  = wmma_load_a(&sh_h[kt * 32], 64, lane);
      v16h bm = wmma_load_b_sw(sh_w, kt * 8 + nt, lane);
      acc = wmma_f16(a, bm, acc);
    }
    int colp = nt * 16 + col15;
#pragma unroll
    for (int i = 0; i < 8; ++i)
      sh_a[(rsel + i) * 128 + colp] = (_Float16)fmaxf(acc[i] + Q_b1[colp], 0.f);
  }

  // Q head layer 2: Qo = t1 @ Q_W2 + Q_b2 (K=128 as two 64x128 chunks)
  v8f q8[8];
#pragma unroll
  for (int nt = 0; nt < 8; ++nt) q8[nt] = vzero;
  for (int kc = 0; kc < 2; ++kc) {
    __syncthreads();
    stage_b<64, 128, 64>(sh_w, Q_W2 + (size_t)kc * 64 * 128, lane, 32);
    __syncthreads();
    for (int nt = 0; nt < 8; ++nt)
#pragma unroll
      for (int kt = 0; kt < 2; ++kt) {
        v16h a  = wmma_load_a(&sh_a[kc * 64 + kt * 32], 128, lane);
        v16h bm = wmma_load_b_sw(sh_w, kt * 8 + nt, lane);
        q8[nt] = wmma_f16(a, bm, q8[nt]);
      }
  }
  for (int nt = 0; nt < 8; ++nt) {
    int colp = nt * 16 + col15;
#pragma unroll
    for (int i = 0; i < 8; ++i)
      sh_qo[(rsel + i) * 128 + colp] = q8[nt][i] + Q_b2[colp];
  }

  // V head layer 1: Vt = relu(hsum @ V_W1 + V_b1)
  __syncthreads();
  stage_b<64, 128, 64>(sh_w, V_W1, lane, 32);
  __syncthreads();
  for (int nt = 0; nt < 8; ++nt) {
    v8f acc = vzero;
#pragma unroll
    for (int kt = 0; kt < 2; ++kt) {
      v16h a  = wmma_load_a(&sh_h[kt * 32], 64, lane);
      v16h bm = wmma_load_b_sw(sh_w, kt * 8 + nt, lane);
      acc = wmma_f16(a, bm, acc);
    }
    int colp = nt * 16 + col15;
#pragma unroll
    for (int i = 0; i < 8; ++i)
      sh_vt[(rsel + i) * 128 + colp] = fmaxf(acc[i] + V_b1[colp], 0.f);
  }
  __syncthreads();
  if (lane < 16) {
    float vo = V_b2[0], qs = 0.f;
    for (int c = 0; c < 128; ++c) {
      vo += sh_vt[lane * 128 + c] * V_W2[c];
      qs += sh_qo[lane * 128 + c];
    }
    sh_vo[lane] = vo;
    sh_qm[lane] = qs * (1.f / 128.f);
  }
  __syncthreads();
  for (int i = lane; i < 16 * 128; i += 32) {
    int r = i >> 7;
    out[(size_t)(b0 + r) * 128 + (i & 127)] = sh_qo[i] - sh_qm[r] + sh_vo[r];
  }
}

extern "C" void kernel_launch(void* const* d_in, const int* in_sizes, int n_in,
                              void* d_out, int out_size, void* d_ws, size_t ws_size,
                              hipStream_t stream) {
  (void)in_sizes; (void)n_in; (void)out_size; (void)ws_size;
  const float* x        = (const float*)d_in[0];
  const float* tw_W1    = (const float*)d_in[1];
  const float* tw_b1    = (const float*)d_in[2];
  const float* tw_gamma = (const float*)d_in[3];
  const float* tw_beta  = (const float*)d_in[4];
  const float* tw_W2    = (const float*)d_in[5];
  const float* tw_b2    = (const float*)d_in[6];
  const float* wq_W1    = (const float*)d_in[7];
  const float* wq_b1    = (const float*)d_in[8];
  const float* wq_W2    = (const float*)d_in[9];
  const float* wq_b2    = (const float*)d_in[10];
  const float* alpha    = (const float*)d_in[11];
  const float* Q_W1     = (const float*)d_in[12];
  const float* Q_b1     = (const float*)d_in[13];
  const float* Q_W2     = (const float*)d_in[14];
  const float* Q_b2     = (const float*)d_in[15];
  const float* V_W1     = (const float*)d_in[16];
  const float* V_b1     = (const float*)d_in[17];
  const float* V_W2     = (const float*)d_in[18];
  const float* V_b2     = (const float*)d_in[19];
  float* out = (float*)d_out;

  // workspace layout (floats): stats[8] | partials[240*8] | qv[1024*64] | r[4*1024*128]
  float* wsf   = (float*)d_ws;
  float* stats = wsf;
  float* part  = wsf + 8;
  float* qvw   = wsf + 2048;
  float* rws   = wsf + 2048 + 65536;

  k_stats_partial<<<240, 256, 0, stream>>>(x, part);
  k_stats_final<<<1, 32, 0, stream>>>(part, stats, 240);
  k_qv<<<BA_N, 64, 0, stream>>>(x, wq_W1, wq_b1, wq_W2, wq_b2, alpha, qvw);
  dim3 g3(BA_N, 4);
  k_attn<<<g3, 256, 0, stream>>>(x, tw_W1, tw_b1, tw_gamma, tw_beta, tw_W2, tw_b2,
                                 stats, qvw, rws);
  k_head<<<64, 32, 0, stream>>>(rws, tw_W2, tw_b2, Q_W1, Q_b1, Q_W2, Q_b2,
                                V_W1, V_b1, V_W2, V_b2, out);
}
